// LocalGlobalAttention_8263517078172
// MI455X (gfx1250) — matrix operand order repaired and checked by reading, not compiled, not run on hardware
//
#include <hip/hip_runtime.h>

// MI455X / gfx1250 implementation of LocalGlobalAttention forward.
//
// Math: softmax row-sums are exactly 1 in the reference einsum
// ('bhqk,bhvd->bhqd' sums k and v independently), so each attention branch is
//   x + broadcast(sum over the 64 V-channels of each head),
// and the V head-sum is a precomputed 64-tap dot of the depthwise conv output
// of input channels 512..767 only. All heavy remaining work (ms pointwise
// convs M=256,K=768 and the final 1x1 conv M=768,K=768, N=pixels) is done as
// f16 WMMA GEMMs (v_wmma_f32_16x16x32_f16) with fp32 accumulation.
//
// v2: pixel-major activations are zero-padded to a multiple of 128 pixels so
// the WMMA B fragments load unconditionally (no per-iteration cndmask/nops),
// and each wave now computes a 64(M)x32(N) tile: 8 WMMAs per 12 b128 loads.

typedef __attribute__((ext_vector_type(16))) _Float16 v16h;
typedef __attribute__((ext_vector_type(8)))  _Float16 v8h;
typedef __attribute__((ext_vector_type(8)))  float    v8f;

namespace lga {
constexpr int Bn  = 32;
constexpr int C   = 768;
constexpr int H   = 28;
constexpr int W   = 28;
constexpr int HW  = 784;
constexpr int NH  = 12;
constexpr int CV  = 256;   // V-relevant channels (512..767)
constexpr int COF = 512;   // channel offset of V groups

static inline int cdiv(int a, int b) { return (a + b - 1) / b; }
static inline int pad128(int n) { return (n + 127) & ~127; }
}

// ---------------------------------------------------------------------------
// WMMA GEMM: Out[b][M][N] = Wm[M][K] * Xt[b][N][K]^T + bias[M]
// Wm f16 row-major [M][K]; Xt f16 pixel-major [Npad][K] (rows >= N are zero);
// Out fp32 [M][N]. M % 64 == 0, K % 32 == 0, Npad % 128 == 0.
// Block = (32,4): 4 waves, each a 64(M)x32(N) tile; block covers 128 N.
// ---------------------------------------------------------------------------
__global__ __launch_bounds__(128)
void lga_gemm_wmma(const _Float16* __restrict__ Wm, const _Float16* __restrict__ Xt,
                   const float* __restrict__ bias, float* __restrict__ Out,
                   int M, int N, int K, long strideXt, long strideOut)
{
  const int lane  = threadIdx.x;        // 0..31
  const int wave  = threadIdx.y;        // 0..3
  const int b     = blockIdx.z;
  const int mBase = blockIdx.y * 64;
  const int nBase = (blockIdx.x * 4 + wave) * 32;

  const int r = lane & 15;              // A row within 16 / B col within 16
  const int g = lane >> 4;              // lane half-group

  const _Float16* Xb = Xt + (long)b * strideXt;
  float*          Ob = Out + (long)b * strideOut;

  const int pix0 = nBase + r;
  const int pix1 = nBase + 16 + r;

  const _Float16* arow = Wm + (long)(mBase + r) * K;   // A: row m = r (+16*mi)
  const _Float16* br0  = Xb + (long)pix0 * K;          // B tile 0: col n = r
  const _Float16* br1  = Xb + (long)pix1 * K;          // B tile 1: col n = r

  v8f acc00 = {}, acc01 = {}, acc10 = {}, acc11 = {};
  v8f acc20 = {}, acc21 = {}, acc30 = {}, acc31 = {};

  for (int k0 = 0; k0 < K; k0 += 32) {
    union U { v16h v; v8h h[2]; };
    // ---- A fragments: halves 0..7 = K[k0+g*8 ..], halves 8..15 = K[k0+16+g*8 ..]
    U ua;
    const _Float16* ap = arow + k0 + g * 8;
    ua.h[0] = *(const v8h*)(ap);
    ua.h[1] = *(const v8h*)(ap + 16);
    v16h a0 = ua.v;
    ua.h[0] = *(const v8h*)(ap + 16 * K);
    ua.h[1] = *(const v8h*)(ap + 16 * K + 16);
    v16h a1 = ua.v;
    ua.h[0] = *(const v8h*)(ap + 32 * K);
    ua.h[1] = *(const v8h*)(ap + 32 * K + 16);
    v16h a2 = ua.v;
    ua.h[0] = *(const v8h*)(ap + 48 * K);
    ua.h[1] = *(const v8h*)(ap + 48 * K + 16);
    v16h a3 = ua.v;
    // ---- B fragments: halves 0..7 = K[k0+g*8 ..], halves 8..15 = K[k0+16+g*8 ..] of the pixel row
    U ub;
    const _Float16* bp0 = br0 + k0 + g * 8;
    ub.h[0] = *(const v8h*)(bp0);
    ub.h[1] = *(const v8h*)(bp0 + 16);
    v16h bf0 = ub.v;
    const _Float16* bp1 = br1 + k0 + g * 8;
    ub.h[0] = *(const v8h*)(bp1);
    ub.h[1] = *(const v8h*)(bp1 + 16);
    v16h bf1 = ub.v;

    acc00 = __builtin_amdgcn_wmma_f32_16x16x32_f16(false, a0, false, bf0, (short)0, acc00, false, false);
    acc10 = __builtin_amdgcn_wmma_f32_16x16x32_f16(false, a1, false, bf0, (short)0, acc10, false, false);
    acc20 = __builtin_amdgcn_wmma_f32_16x16x32_f16(false, a2, false, bf0, (short)0, acc20, false, false);
    acc30 = __builtin_amdgcn_wmma_f32_16x16x32_f16(false, a3, false, bf0, (short)0, acc30, false, false);
    acc01 = __builtin_amdgcn_wmma_f32_16x16x32_f16(false, a0, false, bf1, (short)0, acc01, false, false);
    acc11 = __builtin_amdgcn_wmma_f32_16x16x32_f16(false, a1, false, bf1, (short)0, acc11, false, false);
    acc21 = __builtin_amdgcn_wmma_f32_16x16x32_f16(false, a2, false, bf1, (short)0, acc21, false, false);
    acc31 = __builtin_amdgcn_wmma_f32_16x16x32_f16(false, a3, false, bf1, (short)0, acc31, false, false);
  }

  // D layout: lane (n=r, g), vgpr i -> element (m = g*8+i, n)
  const bool pv0 = (pix0 < N);
  const bool pv1 = (pix1 < N);
  v8f acc0s[4] = {acc00, acc10, acc20, acc30};
  v8f acc1s[4] = {acc01, acc11, acc21, acc31};
#pragma unroll
  for (int mi = 0; mi < 4; ++mi) {
#pragma unroll
    for (int i = 0; i < 8; ++i) {
      const int row = mBase + mi * 16 + g * 8 + i;
      const float bv = bias[row];
      if (pv0) Ob[(long)row * N + pix0] = acc0s[mi][i] + bv;
      if (pv1) Ob[(long)row * N + pix1] = acc1s[mi][i] + bv;
    }
  }
}

// ---------------------------------------------------------------------------
// fp32 -> f16 weight conversion (contiguous)
// ---------------------------------------------------------------------------
__global__ void lga_cvt_f16(const float* __restrict__ src, _Float16* __restrict__ dst, int n)
{
  int i = blockIdx.x * 256 + threadIdx.x;
  if (i < n) dst[i] = (_Float16)src[i];
}

// ---------------------------------------------------------------------------
// f16 NCHW [b][Cn][HWn] -> f16 pixel-major [b][HWpad][Cn], zero-padded rows
// for pix in [HWn, HWpad). Cn % 32 == 0, HWpad % 32 == 0. Block (32,8).
// ---------------------------------------------------------------------------
__global__ __launch_bounds__(256)
void lga_transpose_px(const _Float16* __restrict__ src, _Float16* __restrict__ dst,
                      int Cn, int HWn, int HWpad)
{
  __shared__ _Float16 tile[32][33];
  const int b  = blockIdx.z;
  const int p0 = blockIdx.x * 32;
  const int c0 = blockIdx.y * 32;
  const _Float16* sb = src + (long)b * Cn * HWn;
  _Float16*       db = dst + (long)b * HWn * 0 + (long)b * HWpad * Cn;
  for (int i = threadIdx.y; i < 32; i += 8) {
    const int pix = p0 + threadIdx.x;
    _Float16 v = (_Float16)0.f;
    if (pix < HWn) v = sb[(long)(c0 + i) * HWn + pix];
    tile[i][threadIdx.x] = v;
  }
  __syncthreads();
  for (int i = threadIdx.y; i < 32; i += 8) {
    const int pix = p0 + i;
    if (pix < HWpad) db[(long)pix * Cn + c0 + threadIdx.x] = tile[threadIdx.x][i];
  }
}

// ---------------------------------------------------------------------------
// Generic depthwise conv (NCHW, fp32 in, OutT out). One thread per output.
// ---------------------------------------------------------------------------
template <typename OutT>
__global__ void lga_dwconv(const float* __restrict__ src, const float* __restrict__ wgt,
                           const float* __restrict__ bias, OutT* __restrict__ out,
                           int Ctot, int Cn, int cOffSrc, int cOffW,
                           int Hin, int Win, int k, int stride, int pad,
                           int Hout, int Wout, int total)
{
  int idx = blockIdx.x * 256 + threadIdx.x;
  if (idx >= total) return;
  int t  = idx;
  const int ox = t % Wout; t /= Wout;
  const int oy = t % Hout; t /= Hout;
  const int c  = t % Cn;
  const int b  = t / Cn;
  const float* sp = src + ((long)b * Ctot + cOffSrc + c) * Hin * Win;
  const float* wp = wgt + (long)(cOffW + c) * k * k;
  float acc = bias[cOffW + c];
  const int iy0 = oy * stride - pad, ix0 = ox * stride - pad;
  for (int ky = 0; ky < k; ++ky) {
    const int iy = iy0 + ky;
    if ((unsigned)iy >= (unsigned)Hin) continue;
    for (int kx = 0; kx < k; ++kx) {
      const int ix = ix0 + kx;
      if ((unsigned)ix >= (unsigned)Win) continue;
      acc += wp[ky * k + kx] * sp[iy * Win + ix];
    }
  }
  out[idx] = (OutT)acc;
}

// ---------------------------------------------------------------------------
// Per-branch V head-sum weights: wsum[br][h][j] = sum_t pw[1536+h*64+t][j]
// bsum[br][h] = sum_t pb[1536+h*64+t]. 4*12*64 threads.
// ---------------------------------------------------------------------------
__global__ void lga_wsum(const float* __restrict__ pw0, const float* __restrict__ pw1,
                         const float* __restrict__ pw2, const float* __restrict__ pw3,
                         const float* __restrict__ pb0, const float* __restrict__ pb1,
                         const float* __restrict__ pb2, const float* __restrict__ pb3,
                         float* __restrict__ wsum, float* __restrict__ bsum)
{
  int idx = blockIdx.x * 256 + threadIdx.x;
  if (idx >= 4 * 12 * 64) return;
  const int j  = idx & 63;
  const int h  = (idx >> 6) % 12;
  const int br = idx / (12 * 64);
  const float* pw = (br == 0) ? pw0 : (br == 1) ? pw1 : (br == 2) ? pw2 : pw3;
  const float* pb = (br == 0) ? pb0 : (br == 1) ? pb1 : (br == 2) ? pb2 : pb3;
  const int base = 1536 + h * 64;
  float s = 0.f;
  for (int t = 0; t < 64; ++t) s += pw[(long)(base + t) * 64 + j];
  wsum[idx] = s;
  if (j == 0) {
    float sb = 0.f;
    for (int t = 0; t < 64; ++t) sb += pb[base + t];
    bsum[br * 12 + h] = sb;
  }
}

// ---------------------------------------------------------------------------
// Scale-attention logits + softmax over 4 channels. One thread per pixel.
// ---------------------------------------------------------------------------
__global__ void lga_sa_softmax(const float* __restrict__ x, const float* __restrict__ saw,
                               const float* __restrict__ sab, float* __restrict__ sw)
{
  using namespace lga;
  int t = blockIdx.x * 256 + threadIdx.x;
  if (t >= Bn * HW) return;
  const int b = t / HW, pix = t % HW;
  const float* xp = x + (long)b * C * HW + pix;
  float l0 = sab[0], l1 = sab[1], l2 = sab[2], l3 = sab[3];
  for (int c = 0; c < C; ++c) {
    const float xv = xp[(long)c * HW];
    l0 += saw[c] * xv;
    l1 += saw[C + c] * xv;
    l2 += saw[2 * C + c] * xv;
    l3 += saw[3 * C + c] * xv;
  }
  const float m = fmaxf(fmaxf(l0, l1), fmaxf(l2, l3));
  const float e0 = expf(l0 - m), e1 = expf(l1 - m), e2 = expf(l2 - m), e3 = expf(l3 - m);
  const float inv = 1.f / (e0 + e1 + e2 + e3);
  float* o = sw + (long)b * 4 * HW + pix;
  o[0] = e0 * inv; o[HW] = e1 * inv; o[2 * HW] = e2 * inv; o[3 * HW] = e3 * inv;
}

// ---------------------------------------------------------------------------
// inp[b][c][pix] for c in 512..767 (stored 0..255):
//   sum_i sw_i * feat_i + pos(nearest). feat_i folds nearest-upsample + x.
// ---------------------------------------------------------------------------
__global__ void lga_build_inp(const float* __restrict__ x, const float* __restrict__ ms0,
                              const float* __restrict__ ms1, const float* __restrict__ ms2,
                              const float* __restrict__ sw, const float* __restrict__ pos,
                              float* __restrict__ inp)
{
  using namespace lga;
  int idx = blockIdx.x * 256 + threadIdx.x;
  if (idx >= Bn * CV * HW) return;
  int t = idx;
  const int ox = t % W; t /= W;
  const int oy = t % H; t /= H;
  const int c  = t % CV;
  const int b  = t / CV;
  const int pix = oy * W + ox;
  const float xv = x[((long)b * C + COF + c) * HW + pix];
  const float f0 = xv;
  const float f1 = ms0[((long)b * CV + c) * HW + pix] + xv;
  const float f2 = ms1[((long)b * CV + c) * 196 + (oy >> 1) * 14 + (ox >> 1)] + xv;
  const float f3 = ms2[((long)b * CV + c) * 49 + (oy >> 2) * 7 + (ox >> 2)] + xv;
  const float* sp = sw + (long)b * 4 * HW + pix;
  float v = sp[0] * f0 + sp[HW] * f1 + sp[2 * HW] * f2 + sp[3 * HW] * f3;
  v += pos[(long)(COF + c) * 196 + (oy >> 1) * 14 + (ox >> 1)];
  inp[idx] = v;
}

// ---------------------------------------------------------------------------
// headval: hv[b][h][pix] (+)= bsum[h] + sum_j wsum[h][j]*dwb[b][(h/3)*64+j][pix]
// ---------------------------------------------------------------------------
__global__ void lga_headval(const float* __restrict__ dwb, const float* __restrict__ wsum,
                            const float* __restrict__ bsum, float* __restrict__ hv, int zero)
{
  using namespace lga;
  int idx = blockIdx.x * 256 + threadIdx.x;
  if (idx >= Bn * NH * HW) return;
  int t = idx;
  const int pix = t % HW; t /= HW;
  const int h   = t % NH;
  const int b   = t / NH;
  const float* dp = dwb + ((long)b * CV + (h / 3) * 64) * HW + pix;
  const float* wp = wsum + h * 64;
  float acc = bsum[h];
#pragma unroll 4
  for (int j = 0; j < 64; ++j) acc += wp[j] * dp[(long)j * HW];
  if (zero) hv[idx] = acc; else hv[idx] += acc;
}

// ---------------------------------------------------------------------------
// out_pre = (aL+aG)*x + (aL/3)*hvL[head] + aG*hvG[head]  (f16 NCHW out)
// ---------------------------------------------------------------------------
__global__ void lga_combine(const float* __restrict__ x, const float* __restrict__ hvL,
                            const float* __restrict__ hvG, const float* __restrict__ aL,
                            const float* __restrict__ aG, _Float16* __restrict__ outp)
{
  using namespace lga;
  int idx = blockIdx.x * 256 + threadIdx.x;
  if (idx >= Bn * C * HW) return;
  int t = idx;
  const int pix = t % HW; t /= HW;
  const int c   = t % C;
  const int b   = t / C;
  const float al = *aL, ag = *aG;
  const long hvi = ((long)b * NH + (c >> 6)) * HW + pix;
  const float v = (al + ag) * x[idx] + (al * (1.f / 3.f)) * hvL[hvi] + ag * hvG[hvi];
  outp[idx] = (_Float16)v;
}

// ---------------------------------------------------------------------------
extern "C" void kernel_launch(void* const* d_in, const int* in_sizes, int n_in,
                              void* d_out, int out_size, void* d_ws, size_t ws_size,
                              hipStream_t stream) {
  using namespace lga;
  (void)in_sizes; (void)n_in; (void)out_size; (void)ws_size;

  const float* x = (const float*)d_in[0];
  const float* ms_dw_w[3] = {(const float*)d_in[1], (const float*)d_in[5], (const float*)d_in[9]};
  const float* ms_dw_b[3] = {(const float*)d_in[2], (const float*)d_in[6], (const float*)d_in[10]};
  const float* ms_pw_w[3] = {(const float*)d_in[3], (const float*)d_in[7], (const float*)d_in[11]};
  const float* ms_pw_b[3] = {(const float*)d_in[4], (const float*)d_in[8], (const float*)d_in[12]};
  const float* br_dw_w[4] = {(const float*)d_in[13], (const float*)d_in[17], (const float*)d_in[21], (const float*)d_in[25]};
  const float* br_dw_b[4] = {(const float*)d_in[14], (const float*)d_in[18], (const float*)d_in[22], (const float*)d_in[26]};
  const float* br_pw_w[4] = {(const float*)d_in[15], (const float*)d_in[19], (const float*)d_in[23], (const float*)d_in[27]};
  const float* br_pw_b[4] = {(const float*)d_in[16], (const float*)d_in[20], (const float*)d_in[24], (const float*)d_in[28]};
  const float* pos   = (const float*)d_in[29];
  const float* out_w = (const float*)d_in[30];
  const float* out_b = (const float*)d_in[31];
  const float* sa_w  = (const float*)d_in[32];
  const float* sa_b  = (const float*)d_in[33];
  const float* aL    = (const float*)d_in[34];
  const float* aG    = (const float*)d_in[35];

  const int HWP  = pad128(HW);    // 896
  const int HWP1 = pad128(196);   // 256
  const int HWP2 = pad128(49);    // 128

  // ---- workspace layout (bump allocator, deterministic) ----
  char* ws = (char*)d_ws;
  size_t off = 0;
  auto alloc = [&](size_t bytes) -> char* {
    char* p = ws + off;
    off += (bytes + 255) & ~(size_t)255;
    return p;
  };
  _Float16* dwF16  = (_Float16*)alloc((size_t)Bn * C * HW * 2);    // dw conv f16 / out_pre f16
  _Float16* Xt     = (_Float16*)alloc((size_t)Bn * HWP * C * 2);   // pixel-major f16, zero-padded
  float*    msout0 = (float*)alloc((size_t)Bn * CV * HW * 4);
  float*    msout1 = (float*)alloc((size_t)Bn * CV * 196 * 4);
  float*    msout2 = (float*)alloc((size_t)Bn * CV * 49 * 4);
  float*    sw     = (float*)alloc((size_t)Bn * 4 * HW * 4);
  float*    inp    = (float*)alloc((size_t)Bn * CV * HW * 4);
  float*    dwb    = (float*)alloc((size_t)Bn * CV * HW * 4);
  float*    hvL    = (float*)alloc((size_t)Bn * NH * HW * 4);
  float*    hvG    = (float*)alloc((size_t)Bn * NH * HW * 4);
  float*    wsum   = (float*)alloc(4 * 12 * 64 * 4);
  float*    bsum   = (float*)alloc(4 * 12 * 4);
  _Float16* Wf[3];
  for (int i = 0; i < 3; ++i) Wf[i] = (_Float16*)alloc((size_t)CV * C * 2);
  _Float16* WfO = (_Float16*)alloc((size_t)C * C * 2);
  _Float16* outPre = dwF16;   // alias (dwF16 dead by then)
  _Float16* XtF    = Xt;      // alias

  const dim3 tb(256);

  // ---- weight conversions + head-sum precompute ----
  for (int i = 0; i < 3; ++i) {
    const int n = CV * C;
    lga_cvt_f16<<<cdiv(n, 256), tb, 0, stream>>>(ms_pw_w[i] + (long)COF * C, Wf[i], n);
  }
  lga_cvt_f16<<<cdiv(C * C, 256), tb, 0, stream>>>(out_w, WfO, C * C);
  lga_wsum<<<cdiv(4 * 12 * 64, 256), tb, 0, stream>>>(
      br_pw_w[0], br_pw_w[1], br_pw_w[2], br_pw_w[3],
      br_pw_b[0], br_pw_b[1], br_pw_b[2], br_pw_b[3], wsum, bsum);

  // ---- multi-scale branches: dwconv -> transpose(pad) -> WMMA GEMM (M=256) ----
  float* msout[3] = {msout0, msout1, msout2};
  const int hwp[3] = {HWP, HWP1, HWP2};
  for (int i = 0; i < 3; ++i) {
    const int Ho = H >> i, HWo = Ho * Ho, HWop = hwp[i];
    const int total = Bn * C * HWo;
    lga_dwconv<_Float16><<<cdiv(total, 256), tb, 0, stream>>>(
        x, ms_dw_w[i], ms_dw_b[i], dwF16, C, C, 0, 0, H, W, 3, 1 << i, 1, Ho, Ho, total);
    lga_transpose_px<<<dim3(HWop / 32, C / 32, Bn), dim3(32, 8), 0, stream>>>(
        dwF16, Xt, C, HWo, HWop);
    lga_gemm_wmma<<<dim3(HWop / 128, CV / 64, Bn), dim3(32, 4), 0, stream>>>(
        Wf[i], Xt, ms_pw_b[i] + COF, msout[i], CV, HWo, C, (long)HWop * C, (long)CV * HWo);
  }

  // ---- scale softmax + inp (channels 512..767 only) ----
  lga_sa_softmax<<<cdiv(Bn * HW, 256), tb, 0, stream>>>(x, sa_w, sa_b, sw);
  lga_build_inp<<<cdiv(Bn * CV * HW, 256), tb, 0, stream>>>(x, msout0, msout1, msout2, sw, pos, inp);

  // ---- 4 attention branches, collapsed to dwconv + per-head 64-tap dot ----
  const int ks[4] = {3, 5, 7, 11};
  for (int j = 0; j < 4; ++j) {
    const int k = ks[j];
    const int total = Bn * CV * HW;
    lga_dwconv<float><<<cdiv(total, 256), tb, 0, stream>>>(
        inp, br_dw_w[j], br_dw_b[j], dwb, CV, CV, 0, COF, H, W, k, 1, k / 2, H, W, total);
    lga_headval<<<cdiv(Bn * NH * HW, 256), tb, 0, stream>>>(
        dwb, wsum + j * 12 * 64, bsum + j * 12, (j < 3) ? hvL : hvG, (j == 0 || j == 3) ? 1 : 0);
  }

  // ---- combine + final 1x1 conv (WMMA GEMM M=768) into d_out ----
  lga_combine<<<cdiv(Bn * C * HW, 256), tb, 0, stream>>>(x, hvL, hvG, aL, aG, outPre);
  lga_transpose_px<<<dim3(HWP / 32, C / 32, Bn), dim3(32, 8), 0, stream>>>(outPre, XtF, C, HW, HWP);
  lga_gemm_wmma<<<dim3(HWP / 128, C / 64, Bn), dim3(32, 4), 0, stream>>>(
      WfO, XtF, out_b, (float*)d_out, C, HW, C, (long)HWP * C, (long)C * HW);
}
